// AttentionMixture_22694607192160
// MI455X (gfx1250) — matrix operand order repaired and checked by reading, not compiled, not run on hardware
//
#include <hip/hip_runtime.h>
#include <hip/hip_bf16.h>

typedef _Float16 f16;
typedef __attribute__((ext_vector_type(16))) _Float16 v16h;
typedef __attribute__((ext_vector_type(8)))  _Float16 v8h;
typedef __attribute__((ext_vector_type(4)))  _Float16 v4h;
typedef __attribute__((ext_vector_type(8)))  float    v8f;
typedef __attribute__((ext_vector_type(4)))  float    v4f;

union HF16 { v16h v; v8h h[2]; };

// ---------------------------------------------------------------------------
// WMMA fragment loaders for V_WMMA_F32_16X16X32_F16 (wave32)
// A (16xK=32, 16-bit): lane L -> m = L%16, half = L/16
//   elements 0..7  : K = half*8 + e          (contiguous 8 f16 = 16B)
//   elements 8..15 : K = 16 + half*8 + (e-8) (contiguous 8 f16 = 16B)
// B (K=32 x 16, 16-bit): lane L -> n = L%16, half = L/16
//   elements 0..7  : K = half*8 + e; elements 8..15 : K = 16 + half*8 + (e-8)
// C/D 16x16 f32: lane L -> n = L%16; VGPR r -> m = (L/16)*8 + r
// ---------------------------------------------------------------------------
__device__ __forceinline__ v16h frag_a16(const f16* row, int half) {
  HF16 u;
  u.h[0] = *(const v8h*)(row + half * 8);
  u.h[1] = *(const v8h*)(row + 16 + half * 8);
  return u.v;
}
__device__ __forceinline__ v16h frag_b16(const f16* row, int half) {
  HF16 u;
  u.h[0] = *(const v8h*)(row + half * 8);
  u.h[1] = *(const v8h*)(row + 16 + half * 8);
  return u.v;
}

// ---------------------------------------------------------------------------
// Generic f16 WMMA GEMM: C[M,N] = act(A[M,K] @ B[K,N] + bias)
// block = 256 threads = 8 waves (2x4), block tile 128x256, BK=32,
// per-wave 64x64 (4x4 accumulator frags -> 16 v_wmma per K-step),
// double-buffered LDS (next K-tile staged while current tile computes),
// FLAGS/EXACT compile-time specialized (no runtime branches in hot loop).
// ---------------------------------------------------------------------------
#define BM 128
#define BN 256
#define BKK 32
#define LDT 40   // padded LDS row stride (f16): 80B, conflict-free & 16B aligned

enum { GF_TRANSA = 1, GF_TRANSB = 2, GF_SILU = 4, GF_F16OUT = 8, GF_BIAS = 16 };

template <int FLAGS, bool EXACT>
__global__ __launch_bounds__(256) void gemm_f16_kernel(
    const f16* __restrict__ A, const f16* __restrict__ B,
    const float* __restrict__ bias, void* __restrict__ C,
    int M, int N, int K, int lda, int ldb, int ldc,
    long long sA, long long sB, long long sC)
{
  __shared__ f16 As[2][BM][LDT];
  __shared__ f16 Bs[2][BN][LDT];

  const int tid  = threadIdx.x;
  const int lane = tid & 31;
  const int wid  = tid >> 5;
  const int ln   = lane & 15;
  const int half = lane >> 4;
  const int wm   = wid >> 2;   // 0..1
  const int wn   = wid & 3;    // 0..3
  const int m0   = blockIdx.y * BM;
  const int n0   = blockIdx.x * BN;

  A += (long long)blockIdx.z * sA;
  B += (long long)blockIdx.z * sB;

  const v8f zero8 = {0.f,0.f,0.f,0.f,0.f,0.f,0.f,0.f};
  v8f acc[4][4];
  #pragma unroll
  for (int i = 0; i < 4; ++i)
    #pragma unroll
    for (int j = 0; j < 4; ++j) acc[i][j] = zero8;

  // stage one BKxtile of A and B into LDS buffer `buf`
  auto stage = [&](int k0, int buf) {
    // ---- A tile -> As[buf][m][k]
    if constexpr (!(FLAGS & GF_TRANSA)) {
      #pragma unroll
      for (int it = 0; it < 2; ++it) {
        int id  = it * 256 + tid;
        int row = id >> 2;
        int kc  = (id & 3) << 3;
        const f16* src = A + (long long)(m0 + row) * lda + (k0 + kc);
        v8h val;
        if constexpr (EXACT) {
          val = *(const v8h*)src;
          __builtin_prefetch((const void*)(src + BKK), 0, 0);
        } else {
          int mg = m0 + row;
          if (mg < M && (k0 + kc + 8) <= K) {
            val = *(const v8h*)src;
          } else {
            #pragma unroll
            for (int i = 0; i < 8; ++i)
              val[i] = (mg < M && (k0 + kc + i) < K) ? src[i] : (f16)0.f;
          }
        }
        *(v8h*)&As[buf][row][kc] = val;
      }
    } else {  // A source is [K x M] row-major, transpose on stage
      #pragma unroll
      for (int it = 0; it < 2; ++it) {
        int id = it * 256 + tid;
        int kr = id >> 4;
        int mc = (id & 15) << 3;
        int kg = k0 + kr;
        const f16* src = A + (long long)kg * lda + (m0 + mc);
        f16 tmp[8];
        if constexpr (EXACT) {
          v8h v = *(const v8h*)src;
          #pragma unroll
          for (int i = 0; i < 8; ++i) tmp[i] = v[i];
        } else {
          if (kg < K && (m0 + mc + 8) <= M) {
            v8h v = *(const v8h*)src;
            #pragma unroll
            for (int i = 0; i < 8; ++i) tmp[i] = v[i];
          } else {
            #pragma unroll
            for (int i = 0; i < 8; ++i)
              tmp[i] = (kg < K && (m0 + mc + i) < M) ? src[i] : (f16)0.f;
          }
        }
        #pragma unroll
        for (int i = 0; i < 8; ++i) As[buf][mc + i][kr] = tmp[i];
      }
    }
    // ---- B tile -> Bs[buf][n][k]
    if constexpr (!(FLAGS & GF_TRANSB)) {  // B source is [K x N] row-major
      #pragma unroll
      for (int it = 0; it < 4; ++it) {
        int id = it * 256 + tid;
        int kr = id >> 5;
        int nc = (id & 31) << 3;
        int kg = k0 + kr;
        const f16* src = B + (long long)kg * ldb + (n0 + nc);
        f16 tmp[8];
        if constexpr (EXACT) {
          v8h v = *(const v8h*)src;
          __builtin_prefetch((const void*)(src + (long long)BKK * ldb), 0, 0);
          #pragma unroll
          for (int i = 0; i < 8; ++i) tmp[i] = v[i];
        } else {
          if (kg < K && (n0 + nc + 8) <= N) {
            v8h v = *(const v8h*)src;
            #pragma unroll
            for (int i = 0; i < 8; ++i) tmp[i] = v[i];
          } else {
            #pragma unroll
            for (int i = 0; i < 8; ++i)
              tmp[i] = (kg < K && (n0 + nc + i) < N) ? src[i] : (f16)0.f;
          }
        }
        #pragma unroll
        for (int i = 0; i < 8; ++i) Bs[buf][nc + i][kr] = tmp[i];
      }
    } else {  // B source is [N x K] row-major -> direct vector copy
      #pragma unroll
      for (int it = 0; it < 4; ++it) {
        int id = it * 256 + tid;
        int nr = id >> 2;
        int kc = (id & 3) << 3;
        int ng = n0 + nr;
        const f16* src = B + (long long)ng * ldb + (k0 + kc);
        v8h val;
        if constexpr (EXACT) {
          val = *(const v8h*)src;
          __builtin_prefetch((const void*)(src + BKK), 0, 0);
        } else {
          if (ng < N && (k0 + kc + 8) <= K) {
            val = *(const v8h*)src;
          } else {
            #pragma unroll
            for (int i = 0; i < 8; ++i)
              val[i] = (ng < N && (k0 + kc + i) < K) ? src[i] : (f16)0.f;
          }
        }
        *(v8h*)&Bs[buf][nr][kc] = val;
      }
    }
  };

  const int nk = (K + BKK - 1) / BKK;
  stage(0, 0);
  __syncthreads();

  for (int t = 0; t < nk; ++t) {
    const int cur = t & 1;
    if (t + 1 < nk) stage((t + 1) * BKK, cur ^ 1);  // overlap with WMMA below

    const f16 (*Asc)[LDT] = As[cur];
    const f16 (*Bsc)[LDT] = Bs[cur];
    v16h af[4], bf[4];
    #pragma unroll
    for (int mr = 0; mr < 4; ++mr) af[mr] = frag_a16(&Asc[wm*64 + mr*16 + ln][0], half);
    #pragma unroll
    for (int nr = 0; nr < 4; ++nr) bf[nr] = frag_b16(&Bsc[wn*64 + nr*16 + ln][0], half);
    #pragma unroll
    for (int mr = 0; mr < 4; ++mr)
      #pragma unroll
      for (int nr = 0; nr < 4; ++nr)
        acc[mr][nr] = __builtin_amdgcn_wmma_f32_16x16x32_f16(
            false, af[mr], false, bf[nr], (short)0, acc[mr][nr], false, false);
    __syncthreads();
  }

  // ---- epilogue: bias / silu / store (f16 or f32)
  #pragma unroll
  for (int mr = 0; mr < 4; ++mr) {
    #pragma unroll
    for (int nr = 0; nr < 4; ++nr) {
      int mgb = m0 + wm*64 + mr*16 + half*8;
      int ng  = n0 + wn*64 + nr*16 + ln;
      float bv = 0.f;
      if constexpr ((FLAGS & GF_BIAS) != 0) {
        if (EXACT || ng < N) bv = bias[ng];
      }
      #pragma unroll
      for (int r = 0; r < 8; ++r) {
        int mg = mgb + r;
        if (EXACT || (mg < M && ng < N)) {
          float v = acc[mr][nr][r] + bv;
          if constexpr ((FLAGS & GF_SILU) != 0) v = v / (1.f + __expf(-v));
          long long off = (long long)blockIdx.z * sC + (long long)mg * ldc + ng;
          if constexpr ((FLAGS & GF_F16OUT) != 0) ((f16*)C)[off] = (f16)v;
          else                                    ((float*)C)[off] = v;
        }
      }
    }
  }
}

template <int FLAGS>
static inline void launch_gemm(hipStream_t stream,
    const f16* A, const f16* B, const float* bias, void* C,
    int M, int N, int K, int lda, int ldb, int ldc,
    long long sA, long long sB, long long sC, int batch)
{
  dim3 grid((N + BN - 1) / BN, (M + BM - 1) / BM, batch);
  const bool exact = (M % BM == 0) && (N % BN == 0) && (K % BKK == 0);
  if (exact)
    gemm_f16_kernel<FLAGS, true><<<grid, 256, 0, stream>>>(
        A, B, bias, C, M, N, K, lda, ldb, ldc, sA, sB, sC);
  else
    gemm_f16_kernel<FLAGS, false><<<grid, 256, 0, stream>>>(
        A, B, bias, C, M, N, K, lda, ldb, ldc, sA, sB, sC);
}

// ---------------------------------------------------------------------------
// Fused attention per (b, g, h): S = Q(64x128) @ Kc^T(128x100)/sqrt(128),
// softmax over 100 keys in LDS, Z = Att(64x128pad) @ Vc(100x128).
// block = 128 threads = 4 waves, 64 query rows per block.
// ---------------------------------------------------------------------------
__global__ __launch_bounds__(128) void attn_kernel(
    const f16* __restrict__ q, const f16* __restrict__ kc,
    const f16* __restrict__ vc, f16* __restrict__ z)
{
  __shared__ f16 Vc[128][136];   // [dv][l], l>=100 zero-padded (272B stride: aligned + conflict-free)
  __shared__ f16 Att[64][136];   // scores -> softmaxed att, k-padded to 128 with zeros

  const int tid  = threadIdx.x;
  const int lane = tid & 31;
  const int wid  = tid >> 5;     // 0..3
  const int ln   = lane & 15;
  const int half = lane >> 4;
  const int b    = blockIdx.z;
  const int gh   = blockIdx.y;   // 0..127
  const int g    = gh >> 4;
  const int s0   = blockIdx.x * 64;

  const v8f zero8 = {0.f,0.f,0.f,0.f,0.f,0.f,0.f,0.f};

  // stage Vc transposed: Vc[d][l]  (coalesced across threads per l)
  for (int l = 0; l < 128; ++l) {
    f16 v = (f16)0.f;
    if (l < 100) v = vc[((long long)(b * 100 + l)) * 1024 + g * 128 + tid];
    Vc[tid][l] = v;
  }

  // phase 1: scores = Q @ Kc^T  (Q frags direct from global, Kc frags direct, L2-resident)
  const f16* qrow = q + ((long long)(b * 2048 + s0 + wid * 16 + ln)) * 16384 + (long long)gh * 128;
  v8f sacc[7];
  #pragma unroll
  for (int nf = 0; nf < 7; ++nf) sacc[nf] = zero8;

  for (int k0 = 0; k0 < 128; k0 += 32) {
    HF16 a;
    a.h[0] = *(const v8h*)(qrow + k0 + half * 8);
    a.h[1] = *(const v8h*)(qrow + k0 + 16 + half * 8);
    #pragma unroll
    for (int nf = 0; nf < 7; ++nf) {
      int l = nf * 16 + ln;
      HF16 bfr;
      if (l < 100) {
        const f16* kr = kc + ((long long)(b * 100 + l)) * 1024 + g * 128 + k0 + half * 8;
        bfr.h[0] = *(const v8h*)kr;
        bfr.h[1] = *(const v8h*)(kr + 16);
      } else {
        #pragma unroll
        for (int i = 0; i < 8; ++i) { bfr.h[0][i] = (f16)0.f; bfr.h[1][i] = (f16)0.f; }
      }
      sacc[nf] = __builtin_amdgcn_wmma_f32_16x16x32_f16(
          false, a.v, false, bfr.v, (short)0, sacc[nf], false, false);
    }
  }
  #pragma unroll
  for (int nf = 0; nf < 7; ++nf) {
    int l = nf * 16 + ln;
    if (l < 100) {
      #pragma unroll
      for (int r = 0; r < 8; ++r)
        Att[wid * 16 + half * 8 + r][l] = (f16)(sacc[nf][r] * 0.08838834764831845f);
    }
  }
  __syncthreads();

  // phase 2: row softmax over 100 keys (one thread per query row)
  if (tid < 64) {
    float mx = -1e30f;
    for (int l = 0; l < 100; ++l) mx = fmaxf(mx, (float)Att[tid][l]);
    float sum = 0.f;
    for (int l = 0; l < 100; ++l) {
      float e = __expf((float)Att[tid][l] - mx);
      sum += e;
      Att[tid][l] = (f16)e;
    }
    float inv = 1.f / sum;
    for (int l = 0; l < 100; ++l) Att[tid][l] = (f16)((float)Att[tid][l] * inv);
    for (int l = 100; l < 128; ++l) Att[tid][l] = (f16)0.f;
  }
  __syncthreads();

  // phase 3: Z = Att @ Vc
  v8f zacc[8];
  #pragma unroll
  for (int nf = 0; nf < 8; ++nf) zacc[nf] = zero8;

  for (int k0 = 0; k0 < 128; k0 += 32) {
    HF16 a;
    const f16* ar = &Att[wid * 16 + ln][0];
    a.h[0] = *(const v8h*)(ar + k0 + half * 8);
    a.h[1] = *(const v8h*)(ar + k0 + 16 + half * 8);
    #pragma unroll
    for (int nf = 0; nf < 8; ++nf) {
      HF16 bfr;
      const f16* br = &Vc[nf * 16 + ln][0];
      bfr.h[0] = *(const v8h*)(br + k0 + half * 8);
      bfr.h[1] = *(const v8h*)(br + k0 + 16 + half * 8);
      zacc[nf] = __builtin_amdgcn_wmma_f32_16x16x32_f16(
          false, a.v, false, bfr.v, (short)0, zacc[nf], false, false);
    }
  }
  f16* zr = z + ((long long)(b * 2048 + s0)) * 16384 + (long long)gh * 128;
  #pragma unroll
  for (int nf = 0; nf < 8; ++nf)
    #pragma unroll
    for (int r = 0; r < 8; ++r)
      zr[(long long)(wid * 16 + half * 8 + r) * 16384 + nf * 16 + ln] = (f16)zacc[nf][r];
}

// ---------------------------------------------------------------------------
// l2norm over last dim (vectorized): out = in/max(||in||,1e-12)*sqrt(D)*gamma
// ---------------------------------------------------------------------------
__global__ __launch_bounds__(256) void l2norm_f32_kernel(
    const float* in, const float* gamma, f16* outp, int D, float scale)
{
  __shared__ float red[256];
  const int tid = threadIdx.x;
  const int nc = D >> 2;                    // D % 4 == 0
  const v4f* rp = (const v4f*)(in + (long long)blockIdx.x * D);
  const v4f* gp = (const v4f*)gamma;
  float s = 0.f;
  for (int c = tid; c < nc; c += 256) {
    v4f v = rp[c];
    s += v[0]*v[0] + v[1]*v[1] + v[2]*v[2] + v[3]*v[3];
  }
  red[tid] = s; __syncthreads();
  for (int off = 128; off > 0; off >>= 1) {
    if (tid < off) red[tid] += red[tid + off];
    __syncthreads();
  }
  float r = scale / fmaxf(sqrtf(red[0]), 1e-12f);
  v4h* op = (v4h*)(outp + (long long)blockIdx.x * D);
  for (int c = tid; c < nc; c += 256) {
    v4f v = rp[c], g = gp[c];
    v4h o;
    #pragma unroll
    for (int j = 0; j < 4; ++j) o[j] = (f16)(v[j] * r * g[j]);
    op[c] = o;
  }
}

__global__ __launch_bounds__(256) void l2norm_f16_kernel(
    const f16* in, const float* gamma, f16* outp, int D, float scale)
{
  // D % 8 == 0; in may alias outp (in-place), each element owned by one thread
  __shared__ float red[256];
  const int tid = threadIdx.x;
  const int nc = D >> 3;
  const v8h* rp = (const v8h*)(in + (long long)blockIdx.x * D);
  const v4f* gp = (const v4f*)gamma;
  float s = 0.f;
  for (int c = tid; c < nc; c += 256) {
    v8h v = rp[c];
    #pragma unroll
    for (int j = 0; j < 8; ++j) { float f = (float)v[j]; s += f * f; }
  }
  red[tid] = s; __syncthreads();
  for (int off = 128; off > 0; off >>= 1) {
    if (tid < off) red[tid] += red[tid + off];
    __syncthreads();
  }
  float r = scale / fmaxf(sqrtf(red[0]), 1e-12f);
  v8h* op = (v8h*)(outp + (long long)blockIdx.x * D);
  for (int c = tid; c < nc; c += 256) {
    v8h v = rp[c];
    v4f g0 = gp[2*c], g1 = gp[2*c + 1];
    v8h o;
    #pragma unroll
    for (int j = 0; j < 4; ++j) o[j]     = (f16)((float)v[j]     * r * g0[j]);
    #pragma unroll
    for (int j = 0; j < 4; ++j) o[4 + j] = (f16)((float)v[4 + j] * r * g1[j]);
    op[c] = o;
  }
}

// row softmax over L (per token), scaled by inv_s
__global__ __launch_bounds__(128) void softmax_row_kernel(
    const float* __restrict__ w, float* __restrict__ w1, int ld, int L, float inv_s)
{
  __shared__ float red[128];
  const int tid = threadIdx.x;
  const float* rp = w + (long long)blockIdx.x * ld;
  float mx = -1e30f;
  for (int i = tid; i < L; i += 128) mx = fmaxf(mx, rp[i]);
  red[tid] = mx; __syncthreads();
  for (int off = 64; off > 0; off >>= 1) {
    if (tid < off) red[tid] = fmaxf(red[tid], red[tid + off]);
    __syncthreads();
  }
  float m = red[0]; __syncthreads();
  float s = 0.f;
  for (int i = tid; i < L; i += 128) s += __expf(rp[i] - m);
  red[tid] = s; __syncthreads();
  for (int off = 64; off > 0; off >>= 1) {
    if (tid < off) red[tid] += red[tid + off];
    __syncthreads();
  }
  float tot = red[0];
  float* op = w1 + (long long)blockIdx.x * ld;
  for (int i = tid; i < L; i += 128) op[i] = __expf(rp[i] - m) / tot * inv_s;
}

// column softmax over s (per b,l), combine with w1, write padded f16 mixing weights
__global__ __launch_bounds__(256) void softmax_col_kernel(
    const float* __restrict__ w, const float* __restrict__ w1, f16* __restrict__ wf,
    int S, int ld, int L, float lsw)
{
  const int tid = threadIdx.x;
  const int l = blockIdx.x, b = blockIdx.y;
  const long long base = (long long)b * S * ld + l;
  if (l >= L) {   // zero padding columns (uniform branch per block)
    for (int s = tid; s < S; s += 256) wf[base + (long long)s * ld] = (f16)0.f;
    return;
  }
  __shared__ float red[256];
  float mx = -1e30f;
  for (int s = tid; s < S; s += 256) mx = fmaxf(mx, w[base + (long long)s * ld]);
  red[tid] = mx; __syncthreads();
  for (int off = 128; off > 0; off >>= 1) {
    if (tid < off) red[tid] = fmaxf(red[tid], red[tid + off]);
    __syncthreads();
  }
  float m = red[0]; __syncthreads();
  float sum = 0.f;
  for (int s = tid; s < S; s += 256) sum += __expf(w[base + (long long)s * ld] - m);
  red[tid] = sum; __syncthreads();
  for (int off = 128; off > 0; off >>= 1) {
    if (tid < off) red[tid] += red[tid + off];
    __syncthreads();
  }
  float inv2 = lsw / ((float)L * red[0]);
  float c = 1.f / (1.f + lsw);
  for (int s = tid; s < S; s += 256) {
    long long o = base + (long long)s * ld;
    float v2 = __expf(w[o] - m) * inv2;
    wf[o] = (f16)((w1[o] + v2) * c);
  }
}

// vectorized fp32 -> f16 (n % 8 == 0), n8 = n/8
__global__ void cvt_kernel(const float* __restrict__ s, f16* __restrict__ d, long long n8) {
  long long i = (long long)blockIdx.x * blockDim.x + threadIdx.x;
  if (i < n8) {
    const v4f* sp = (const v4f*)s + i * 2;
    v4f a = sp[0], b = sp[1];
    v8h o;
    #pragma unroll
    for (int j = 0; j < 4; ++j) { o[j] = (f16)a[j]; o[4 + j] = (f16)b[j]; }
    ((v8h*)d)[i] = o;
  }
}
__global__ void cvt_pad_kernel(const float* __restrict__ s, f16* __restrict__ d,
                               int rows, int cols, int ldd) {
  long long i = (long long)blockIdx.x * blockDim.x + threadIdx.x;
  long long tot = (long long)rows * ldd;
  if (i < tot) {
    int r = (int)(i / ldd), c = (int)(i % ldd);
    d[i] = (c < cols) ? (f16)s[(long long)r * cols + c] : (f16)0.f;
  }
}

// ---------------------------------------------------------------------------
extern "C" void kernel_launch(void* const* d_in, const int* in_sizes, int n_in,
                              void* d_out, int out_size, void* d_ws, size_t ws_size,
                              hipStream_t stream) {
  const float* x     = (const float*)d_in[0];
  const float* gamma = (const float*)d_in[1];
  const float* Wq_w  = (const float*)d_in[2];
  const float* Wq_b  = (const float*)d_in[3];
  const float* Wkv_w = (const float*)d_in[4];
  const float* Wkv_b = (const float*)d_in[5];
  const float* aw1_w = (const float*)d_in[6];
  const float* aw1_b = (const float*)d_in[7];
  const float* aw2_w = (const float*)d_in[8];
  const float* aw2_b = (const float*)d_in[9];
  const float* wz_g1 = (const float*)d_in[10];
  const float* wz_w1 = (const float*)d_in[11];
  const float* wz_b1 = (const float*)d_in[12];
  const float* wz_g2 = (const float*)d_in[13];
  const float* wz_w2 = (const float*)d_in[14];
  const float* wz_b2 = (const float*)d_in[15];
  float* out = (float*)d_out;

  // ---- carve workspace (256B-aligned)
  char* p = (char*)d_ws;
  auto carve = [&p](size_t bytes) -> char* {
    char* r = p;
    p += (bytes + 255) & ~(size_t)255;
    return r;
  };
  f16*   Wq_h   = (f16*)  carve((size_t)512 * 16384 * 2);
  f16*   Wkv_h  = (f16*)  carve((size_t)512 * 2048 * 2);
  f16*   aw1_h  = (f16*)  carve((size_t)1024 * 112 * 2);
  f16*   aw2_h  = (f16*)  carve((size_t)100 * 112 * 2);
  f16*   wz1_h  = (f16*)  carve((size_t)16384 * 1024 * 2);
  f16*   wz2_h  = (f16*)  carve((size_t)1024 * 512 * 2);
  f16*   xn_h   = (f16*)  carve((size_t)8192 * 512 * 2);
  f16*   q_h    = (f16*)  carve((size_t)8192 * 16384 * 2);
  f16*   kv_h   = (f16*)  carve((size_t)8192 * 2048 * 2);
  f16*   wpre_h = (f16*)  carve((size_t)8192 * 112 * 2);
  float* w_f32  = (float*)carve((size_t)8192 * 112 * 4);
  float* w1_f32 = (float*)carve((size_t)8192 * 112 * 4);
  f16*   wf_h   = (f16*)  carve((size_t)8192 * 112 * 2);
  f16*   kc_h   = (f16*)  carve((size_t)4 * 100 * 1024 * 2);
  f16*   vc_h   = (f16*)  carve((size_t)4 * 100 * 1024 * 2);
  f16*   z_h    = (f16*)  carve((size_t)8192 * 16384 * 2);
  f16*   h1_h   = (f16*)  carve((size_t)8192 * 1024 * 2);

  auto cvt = [&](const float* s, f16* d, long long n) {
    long long n8 = n >> 3;
    cvt_kernel<<<(unsigned)((n8 + 255) / 256), 256, 0, stream>>>(s, d, n8);
  };

  // weight conversions (fp32 -> f16, L2-resident afterwards: 192MB L2 holds all)
  cvt(Wq_w,  Wq_h,  (long long)512 * 16384);
  cvt(Wkv_w, Wkv_h, (long long)512 * 2048);
  cvt(wz_w1, wz1_h, (long long)16384 * 1024);
  cvt(wz_w2, wz2_h, (long long)1024 * 512);
  cvt_pad_kernel<<<(1024 * 112 + 255) / 256, 256, 0, stream>>>(aw1_w, aw1_h, 1024, 100, 112);
  cvt_pad_kernel<<<(100 * 112 + 255) / 256, 256, 0, stream>>>(aw2_w, aw2_h, 100, 100, 112);

  // xn = l2norm(x)*gamma  (sqrt(512))
  l2norm_f32_kernel<<<8192, 256, 0, stream>>>(x, gamma, xn_h, 512, 22.62741699796952f);

  // q = xn @ Wq + b   (8192x16384, K=512)  [exact path]
  launch_gemm<GF_F16OUT | GF_BIAS>(stream, xn_h, Wq_h, Wq_b, q_h,
      8192, 16384, 512, 512, 16384, 16384, 0, 0, 0, 1);
  // kv = xn @ Wkv + b (8192x2048, K=512); k = cols[0:1024), v = cols[1024:2048)
  launch_gemm<GF_F16OUT | GF_BIAS>(stream, xn_h, Wkv_h, Wkv_b, kv_h,
      8192, 2048, 512, 512, 2048, 2048, 0, 0, 0, 1);

  // address MLP: wpre = silu(k @ aw1 + b), w = wpre @ aw2 + b (f32)
  launch_gemm<GF_F16OUT | GF_BIAS | GF_SILU>(stream, kv_h, aw1_h, aw1_b, wpre_h,
      8192, 100, 1024, 2048, 112, 112, 0, 0, 0, 1);
  launch_gemm<GF_BIAS>(stream, wpre_h, aw2_h, aw2_b, w_f32,
      8192, 100, 100, 112, 112, 112, 0, 0, 0, 1);

  // dual softmax + combine -> f16 mixing weights wf[b,s,112]
  softmax_row_kernel<<<8192, 128, 0, stream>>>(w_f32, w1_f32, 112, 100, 1.f / 2048.f);
  softmax_col_kernel<<<dim3(112, 4), 256, 0, stream>>>(w_f32, w1_f32, wf_h, 2048, 112, 100, 1.0f);

  // kc[b,l,d] = sum_s w[s,l]*k[s,d]  -> batched GEMM, A = wf (transA), B = k
  launch_gemm<GF_TRANSA | GF_F16OUT>(stream, wf_h, kv_h, nullptr, kc_h,
      100, 1024, 2048, 112, 2048, 1024,
      (long long)2048 * 112, (long long)2048 * 2048, (long long)100 * 1024, 4);
  launch_gemm<GF_TRANSA | GF_F16OUT>(stream, wf_h, kv_h + 1024, nullptr, vc_h,
      100, 1024, 2048, 112, 2048, 1024,
      (long long)2048 * 112, (long long)2048 * 2048, (long long)100 * 1024, 4);

  // fused attention -> z (f16)
  attn_kernel<<<dim3(32, 128, 4), 128, 0, stream>>>(q_h, kc_h, vc_h, z_h);

  // zn = l2norm(z)*g1 (in place, sqrt(16384)=128)
  l2norm_f16_kernel<<<8192, 256, 0, stream>>>(z_h, wz_g1, z_h, 16384, 128.f);

  // h1 = silu(zn @ wz1 + b1)  (8192x1024, K=16384 -- the big one) [exact path]
  launch_gemm<GF_F16OUT | GF_BIAS | GF_SILU>(stream, z_h, wz1_h, wz_b1, h1_h,
      8192, 1024, 16384, 16384, 1024, 1024, 0, 0, 0, 1);

  // h1n = l2norm(h1)*g2 (in place, sqrt(1024)=32)
  l2norm_f16_kernel<<<8192, 256, 0, stream>>>(h1_h, wz_g2, h1_h, 1024, 32.f);

  // out = silu(h1n @ wz2 + b2) -> fp32 d_out [exact path]
  launch_gemm<GF_BIAS | GF_SILU>(stream, h1_h, wz2_h, wz_b2, out,
      8192, 512, 1024, 1024, 512, 512, 0, 0, 0, 1);
}
